// Planner_70781061038118
// MI455X (gfx1250) — matrix operand order repaired and checked by reading, not compiled, not run on hardware
//
#include <hip/hip_runtime.h>

// ---------------------------------------------------------------------------
// Problem constants (from reference): B=32, ORIENT=4, ACTIONS=8, L_H=128,
// K=20, F=3, M=32.  Flat pixel layout: n = (b*32 + w)*32 + h, channels last.
// Padded layout for conv input: [B][34][34][128], interior at (w+1, h+1).
// ---------------------------------------------------------------------------
#define NPIX   32768      // B*M*M
#define LH     128
#define MSZ    32
#define MP     34         // padded spatial
#define KSTEPS 19         // K-1
#define LDSTR  136        // LDS row stride in halves (272B = 68 dwords -> 4-bank skew)

typedef __attribute__((ext_vector_type(16))) _Float16 v16h;
typedef __attribute__((ext_vector_type(8)))  float    v8f;

union FragH { v16h v; unsigned u[8]; _Float16 h[16]; };

// A fragment: 16x32 f16 tile from a row base (global or LDS), unconditional.
// Lane = row (lanes 16-31 duplicate rows, carry the other K half).
// VGPR v<4  -> K = (hi?16:0) + 2v,2v+1
// VGPR v>=4 -> K = (hi?24:8) + 2(v-4),...   (contiguous K = 16*hi + s, matches frag_b)
__device__ inline v16h frag_a(const _Float16* rowp, int kb, int hi) {
  FragH f;
  const unsigned* p = (const unsigned*)(rowp + kb);
  const int o1 = hi ? 8 : 0;
  const int o2 = hi ? 12 : 4;
#pragma unroll
  for (int v = 0; v < 4; ++v) f.u[v]     = p[o1 + v];
#pragma unroll
  for (int v = 0; v < 4; ++v) f.u[4 + v] = p[o2 + v];
  return f.v;
}

// B fragment: 32x16 f16 tile from an N-major buffer (col*128 + k).
// Lanes 0-15: K=kb+0..15, lanes 16-31: K=kb+16..31, col = lane&15.
__device__ inline v16h frag_b(const _Float16* colp, int kb, int hi) {
  FragH f;
  const unsigned* p = (const unsigned*)(colp + kb + (hi ? 16 : 0));
#pragma unroll
  for (int v = 0; v < 8; ++v) f.u[v] = p[v];
  return f.v;
}

__device__ inline v8f wmma16(v16h a, v16h b, v8f c) {
  return __builtin_amdgcn_wmma_f32_16x16x32_f16(
      /*neg_a=*/false, a, /*neg_b=*/false, b,
      /*c_mod=*/(short)0, c, /*reuse_a=*/false, /*reuse_b=*/false);
}

__device__ inline float sigm(float x) { return 1.f / (1.f + expf(-x)); }

// ---------------------------------------------------------------------------
// Weight / buffer prep kernels
// ---------------------------------------------------------------------------
__global__ void prep_f32_to_f16(const float* __restrict__ s,
                                _Float16* __restrict__ d, int n) {
  int i = blockIdx.x * blockDim.x + threadIdx.x;
  if (i < n) d[i] = (_Float16)s[i];
}

// [O=128][I=128][3][3] f32 -> [tap][cout][cin] f16 (N-major per tap)
__global__ void prep_conv9(const float* __restrict__ w,
                           _Float16* __restrict__ d) {
  int i = blockIdx.x * blockDim.x + threadIdx.x;
  if (i >= 9 * 128 * 128) return;
  int ci = i & 127, co = (i >> 7) & 127, t = i >> 14;
  d[i] = (_Float16)w[(co * 128 + ci) * 9 + t];
}

// conv_w [1,128,3,3] -> [tap][c] f32 (contiguous per tap for float4 dots)
__global__ void prep_convw(const float* __restrict__ w,
                           float* __restrict__ d) {
  int i = blockIdx.x * blockDim.x + threadIdx.x;   // 9*128
  if (i >= 9 * 128) return;
  int c = i & 127, t = i >> 7;
  d[t * 128 + c] = w[c * 9 + t];
}

__global__ void zero_f16_vec(_Float16* __restrict__ p, int n16) {
  int i = blockIdx.x * blockDim.x + threadIdx.x;
  if (i < n16) { uint4 z = {0u, 0u, 0u, 0u}; ((uint4*)p)[i] = z; }
}

// ---------------------------------------------------------------------------
// Input conv: hid = conv3x3(concat(design, goal)) + bias (5 -> 128),
// written into the ZERO-PADDED channels-last f16 buffer at (w+1, h+1).
// ---------------------------------------------------------------------------
__global__ void hid_conv_kernel(const float* __restrict__ design,
                                const float* __restrict__ goal,
                                const float* __restrict__ hw,
                                const float* __restrict__ hb,
                                _Float16* __restrict__ hidP) {
  int idx = blockIdx.x * blockDim.x + threadIdx.x;   // NPIX*128
  if (idx >= NPIX * LH) return;
  int co = idx & 127, p = idx >> 7;
  int b = p >> 10, rw = (p >> 5) & 31, rh = p & 31;
  float s = hb[co];
#pragma unroll
  for (int t = 0; t < 9; ++t) {
    int dh = t / 3 - 1, dw = t % 3 - 1;
    int nw = rw + dw, nh = rh + dh;
    if (nw < 0 || nw > 31 || nh < 0 || nh > 31) continue;
    const float* wp = hw + co * 45 + t;              // + c*9
    float acc = 0.f;
#pragma unroll
    for (int c = 0; c < 4; ++c)
      acc += design[((b * 4 + c) * MSZ + nh) * MSZ + nw] * wp[c * 9];
    acc += goal[(b * MSZ + nh) * MSZ + nw] * wp[4 * 9];
    s += acc;
  }
  hidP[((size_t)(b * MP + rw + 1) * MP + rh + 1) * LH + co] = (_Float16)s;
}

// ---------------------------------------------------------------------------
// 3x3 conv as 9-tap implicit GEMM with WMMA (128 -> 128), K=128 per tap.
// Input is zero-padded -> unconditional A loads. The 3x18-row halo tile is
// staged once per block into LDS (global_load_b128 -> ds_store_b128) and
// shared by all 8 column-tile waves; A fragments come from ds_load.
// Grid: 2048 blocks x 256 threads.
// ---------------------------------------------------------------------------
__global__ void conv3x3_gemm_wmma(const _Float16* __restrict__ inP,  // padded
                                  const _Float16* __restrict__ w9,   // [9][co][ci]
                                  const float* __restrict__ bias,    // [128]
                                  float* outA, float* outB,
                                  _Float16* out16) {
  __shared__ __align__(16) _Float16 ldsA[3 * 18 * LDSTR];   // ~14.7 KB

  int lane = threadIdx.x & 31, wave = threadIdx.x >> 5;
  int hl = lane & 15, hi = lane >> 4;
  int m0 = blockIdx.x * 16, n0 = wave * 16;
  int b = m0 >> 10, w = (m0 >> 5) & 31, h0 = m0 & 31;      // h0 in {0,16}

  // Stage 3 columns x 18 rows x 128 halves (16B chunks). 864 chunks total.
  for (int i = threadIdx.x; i < 3 * 18 * 16; i += 256) {
    int c3 = i / 288, rem = i - c3 * 288;
    int r = rem >> 4, q = rem & 15;
    const uint4* src = (const uint4*)(inP +
        ((size_t)(b * MP + w + c3) * MP + (h0 + r)) * LH);
    *((uint4*)(ldsA + (size_t)(c3 * 18 + r) * LDSTR) + q) = src[q];
  }
  __syncthreads();

  v8f acc = {};
#pragma unroll
  for (int t = 0; t < 9; ++t) {
    int dh = t / 3 - 1, dw = t % 3 - 1;
    const _Float16* rowp = ldsA + (size_t)((dw + 1) * 18 + hl + dh + 1) * LDSTR;
    const _Float16* colp = w9 + (size_t)(t * 128 + n0 + hl) * LH;
#pragma unroll
    for (int kb = 0; kb < 128; kb += 32) {
      v16h a  = frag_a(rowp, kb, hi);
      v16h bb = frag_b(colp, kb, hi);
      acc = wmma16(a, bb, acc);
    }
  }
  int col = n0 + hl;
  float bv = bias[col];
#pragma unroll
  for (int r = 0; r < 8; ++r) {
    int orow = m0 + r + 8 * hi;
    float v = acc[r] + bv;
    if (outA)  outA[(size_t)orow * LH + col] = v;
    if (outB)  outB[(size_t)orow * LH + col] = v;
    if (out16) out16[(size_t)orow * LH + col] = (_Float16)v;
  }
}

// ---------------------------------------------------------------------------
// Per-step 128->1 conv3x3:  inp[p] = conv_b + sum_{tap,c} h[p',c]*cwT[t,c]
// float4-vectorized dot products.
// ---------------------------------------------------------------------------
__global__ void inp_conv_kernel(const float* __restrict__ h32,
                                const float* __restrict__ cwT,  // [9][128]
                                const float* __restrict__ cb,   // [1]
                                float* __restrict__ inp) {
  int p = blockIdx.x * blockDim.x + threadIdx.x;
  if (p >= NPIX) return;
  int b = p >> 10, rw = (p >> 5) & 31, rh = p & 31;
  float s = cb[0];
#pragma unroll
  for (int t = 0; t < 9; ++t) {
    int dh = t / 3 - 1, dw = t % 3 - 1;
    int nw = rw + dw, nh = rh + dh;
    if (nw < 0 || nw > 31 || nh < 0 || nh > 31) continue;
    const float4* hp4 = (const float4*)(h32 +
        (size_t)((((b << 5) + nw) << 5) + nh) * LH);
    const float4* wp4 = (const float4*)(cwT + t * 128);
    float acc = 0.f;
#pragma unroll
    for (int c4 = 0; c4 < 32; ++c4) {
      float4 a = hp4[c4], ww = wp4[c4];
      acc += a.x * ww.x + a.y * ww.y + a.z * ww.z + a.w * ww.w;
    }
    s += acc;
  }
  inp[p] = s;
}

// ---------------------------------------------------------------------------
// Fused gates-GEMM + LSTM pointwise update.  One wave owns a 16x16 tile of
// all four gate blocks (16 WMMAs reusing 4 A fragments).  The 16 h rows are
// staged once per block into LDS and shared by all 8 column waves.
// Ping-pong f16 h avoids cross-wave RAW; c/h32 are single-writer in place.
// ---------------------------------------------------------------------------
__global__ void lstm_step_wmma(const _Float16* __restrict__ hprev16, // [N,128]
                               const _Float16* __restrict__ whh16,   // [512][128]
                               const float* __restrict__ inp,        // [N]
                               const float* __restrict__ w_ih,       // [512]
                               const float* __restrict__ b_ih,       // [512]
                               const float* __restrict__ b_hh,       // [512]
                               float* __restrict__ c32,              // [N,128]
                               float* __restrict__ h32,              // [N,128]
                               _Float16* __restrict__ hnext16) {     // [N,128]
  __shared__ __align__(16) _Float16 ldsH[16 * LDSTR];   // ~4.3 KB

  int lane = threadIdx.x & 31, wave = threadIdx.x >> 5;
  int hl = lane & 15, hi = lane >> 4;
  int m0 = blockIdx.x * 16, j0 = wave * 16;

  {  // stage 16 rows x 128 halves: one 16B chunk per thread
    int r = threadIdx.x >> 4, q = threadIdx.x & 15;
    const uint4* src = (const uint4*)(hprev16 + (size_t)(m0 + r) * LH);
    *((uint4*)(ldsH + (size_t)r * LDSTR) + q) = src[q];
  }
  __syncthreads();

  const _Float16* rowp = ldsH + (size_t)hl * LDSTR;
  v16h a[4];
#pragma unroll
  for (int kk = 0; kk < 4; ++kk) a[kk] = frag_a(rowp, kk * 32, hi);

  v8f z = {};
  v8f acc[4] = {z, z, z, z};
#pragma unroll
  for (int g = 0; g < 4; ++g) {
    const _Float16* colp = whh16 + (size_t)(g * 128 + j0 + hl) * LH;
#pragma unroll
    for (int kk = 0; kk < 4; ++kk) {
      v16h bb = frag_b(colp, kk * 32, hi);
      acc[g] = wmma16(a[kk], bb, acc[g]);
    }
  }

  int col = j0 + hl;
  float wi_i = w_ih[col      ], bb_i = b_ih[col      ] + b_hh[col      ];
  float wi_f = w_ih[col + 128], bb_f = b_ih[col + 128] + b_hh[col + 128];
  float wi_g = w_ih[col + 256], bb_g = b_ih[col + 256] + b_hh[col + 256];
  float wi_o = w_ih[col + 384], bb_o = b_ih[col + 384] + b_hh[col + 384];

#pragma unroll
  for (int r = 0; r < 8; ++r) {
    int orow = m0 + r + 8 * hi;
    float x  = inp[orow];
    float gi = acc[0][r] + x * wi_i + bb_i;
    float gf = acc[1][r] + x * wi_f + bb_f;
    float gg = acc[2][r] + x * wi_g + bb_g;
    float go = acc[3][r] + x * wi_o + bb_o;
    size_t idx = (size_t)orow * LH + col;
    float cc = sigm(gf) * c32[idx] + sigm(gi) * tanhf(gg);
    float hh = sigm(go) * tanhf(cc);
    c32[idx] = cc;
    h32[idx] = hh;
    hnext16[idx] = (_Float16)hh;
  }
}

// ---------------------------------------------------------------------------
// Policy 1x1 conv: logits[N,32] = h[N,128] @ pol^T[128,32]  (WMMA, K=128).
// Grid: 2048 blocks x 64 threads (2 col-tile waves sharing LDS-staged A).
// ---------------------------------------------------------------------------
__global__ void policy_gemm_wmma(const _Float16* __restrict__ h16,
                                 const _Float16* __restrict__ pol16, // [32][128]
                                 float* __restrict__ logits) {
  __shared__ __align__(16) _Float16 ldsH[16 * LDSTR];

  int lane = threadIdx.x & 31, wave = threadIdx.x >> 5;
  int hl = lane & 15, hi = lane >> 4;
  int m0 = blockIdx.x * 16, n0 = wave * 16;

  for (int i = threadIdx.x; i < 256; i += 64) {
    int r = i >> 4, q = i & 15;
    const uint4* src = (const uint4*)(h16 + (size_t)(m0 + r) * LH);
    *((uint4*)(ldsH + (size_t)r * LDSTR) + q) = src[q];
  }
  __syncthreads();

  const _Float16* rowp = ldsH + (size_t)hl * LDSTR;
  const _Float16* colp = pol16 + (size_t)(n0 + hl) * LH;
  v8f acc = {};
#pragma unroll
  for (int kk = 0; kk < 4; ++kk) {
    v16h a  = frag_a(rowp, kk * 32, hi);
    v16h bb = frag_b(colp, kk * 32, hi);
    acc = wmma16(a, bb, acc);
  }
  int col = n0 + hl;
#pragma unroll
  for (int r = 0; r < 8; ++r)
    logits[(size_t)(m0 + r + 8 * hi) * 32 + col] = acc[r];
}

// ---------------------------------------------------------------------------
// Softmax over actions + layout: [N,32]=[b,w,h][orient*8+a] -> [B,8,4,32,32]
// ---------------------------------------------------------------------------
__global__ void softmax_out_kernel(const float* __restrict__ logits,
                                   float* __restrict__ out_logits,
                                   float* __restrict__ out_probs) {
  int idx = blockIdx.x * blockDim.x + threadIdx.x;   // 32*4*1024
  if (idx >= 32 * 4 * 1024) return;
  int ww = idx & 31, hh = (idx >> 5) & 31, ori = (idx >> 10) & 3, b = idx >> 12;
  int n = ((b << 5) + ww) * 32 + hh;
  float v[8], mx = -1e30f;
#pragma unroll
  for (int a = 0; a < 8; ++a) {
    v[a] = logits[(size_t)n * 32 + ori * 8 + a];
    mx = fmaxf(mx, v[a]);
  }
  float e[8], sum = 0.f;
#pragma unroll
  for (int a = 0; a < 8; ++a) { e[a] = expf(v[a] - mx); sum += e[a]; }
  float inv = 1.f / sum;
#pragma unroll
  for (int a = 0; a < 8; ++a) {
    size_t o = ((size_t)(((b * 8 + a) * 4 + ori)) * 32 + hh) * 32 + ww;
    out_logits[o] = v[a];
    out_probs[o]  = e[a] * inv;
  }
}

// hk[b,c,h,w] = h32[(b*32+w)*32+h, c]
__global__ void hk_out_kernel(const float* __restrict__ h32,
                              float* __restrict__ out) {
  int idx = blockIdx.x * blockDim.x + threadIdx.x;   // 32*128*1024
  if (idx >= 32 * 128 * 1024) return;
  int ww = idx & 31, hh = (idx >> 5) & 31, c = (idx >> 10) & 127, b = idx >> 17;
  out[idx] = h32[(size_t)((((b << 5) + ww) << 5) + hh) * LH + c];
}

// ---------------------------------------------------------------------------
extern "C" void kernel_launch(void* const* d_in, const int* in_sizes, int n_in,
                              void* d_out, int out_size, void* d_ws,
                              size_t ws_size, hipStream_t stream) {
  (void)in_sizes; (void)n_in; (void)out_size; (void)ws_size;
  const float* design = (const float*)d_in[0];
  const float* goal   = (const float*)d_in[1];
  const float* hid_w  = (const float*)d_in[2];
  const float* hid_b  = (const float*)d_in[3];
  const float* h0_w   = (const float*)d_in[4];
  const float* h0_b   = (const float*)d_in[5];
  const float* c0_w   = (const float*)d_in[6];
  const float* c0_b   = (const float*)d_in[7];
  const float* conv_w = (const float*)d_in[8];
  const float* conv_b = (const float*)d_in[9];
  const float* w_ih   = (const float*)d_in[10];
  const float* w_hh   = (const float*)d_in[11];
  const float* b_ih   = (const float*)d_in[12];
  const float* b_hh   = (const float*)d_in[13];
  const float* pol_w  = (const float*)d_in[14];

  float* out_f = (float*)d_out;
  float* out_logits = out_f;                 // 1,048,576
  float* out_probs  = out_f + 1048576;       // 1,048,576
  float* out_h0     = out_f + 2097152;       // 4,194,304
  float* out_hk     = out_f + 6291456;       // 4,194,304

  char* ws = (char*)d_ws;
  size_t off = 0;
  auto carve = [&](size_t bytes) -> void* {
    void* p = ws + off;
    off += (bytes + 255) & ~(size_t)255;
    return p;
  };
  const size_t NPAD16 = (size_t)32 * MP * MP * LH;           // padded hid halves
  _Float16* hidP   = (_Float16*)carve(NPAD16 * 2);
  _Float16* h16a   = (_Float16*)carve((size_t)NPIX * LH * 2);
  _Float16* h16b   = (_Float16*)carve((size_t)NPIX * LH * 2);
  float*    h32    = (float*)   carve((size_t)NPIX * LH * 4);
  float*    c32    = (float*)   carve((size_t)NPIX * LH * 4);
  float*    inp32  = (float*)   carve((size_t)NPIX * 4);
  _Float16* whh16  = (_Float16*)carve((size_t)512 * 128 * 2);
  _Float16* w9h16  = (_Float16*)carve((size_t)9 * 128 * 128 * 2);
  _Float16* w9c16  = (_Float16*)carve((size_t)9 * 128 * 128 * 2);
  _Float16* pol16  = (_Float16*)carve((size_t)32 * 128 * 2);
  float*    cwT    = (float*)   carve((size_t)9 * 128 * 4);
  float*    lg32   = (float*)   carve((size_t)NPIX * 32 * 4);

  // Weight repacks (B matrices N-major f16) + zero the padded hid buffer
  prep_f32_to_f16<<<256, 256, 0, stream>>>(w_hh, whh16, 512 * 128);
  prep_f32_to_f16<<<16, 256, 0, stream>>>(pol_w, pol16, 32 * 128);
  prep_conv9<<<576, 256, 0, stream>>>(h0_w, w9h16);
  prep_conv9<<<576, 256, 0, stream>>>(c0_w, w9c16);
  prep_convw<<<5, 256, 0, stream>>>(conv_w, cwT);
  {
    int n16 = (int)(NPAD16 / 8);                 // 16B chunks
    zero_f16_vec<<<(n16 + 255) / 256, 256, 0, stream>>>(hidP, n16);
  }

  // hid = conv3x3(X) (5->128) into zero-padded channels-last f16
  hid_conv_kernel<<<16384, 256, 0, stream>>>(design, goal, hid_w, hid_b, hidP);

  // h0 / c0 = conv3x3(hid) via LDS-staged 9-tap implicit WMMA GEMM
  conv3x3_gemm_wmma<<<2048, 256, 0, stream>>>(hidP, w9h16, h0_b,
                                              h32, out_h0, h16a);
  conv3x3_gemm_wmma<<<2048, 256, 0, stream>>>(hidP, w9c16, c0_b,
                                              c32, nullptr, nullptr);

  // 19 recurrent steps: inp conv then fused gates-GEMM + LSTM update
  _Float16* hp = h16a;
  _Float16* hn = h16b;
  for (int s = 0; s < KSTEPS; ++s) {
    inp_conv_kernel<<<128, 256, 0, stream>>>(h32, cwT, conv_b, inp32);
    lstm_step_wmma<<<2048, 256, 0, stream>>>(hp, whh16, inp32, w_ih, b_ih,
                                             b_hh, c32, h32, hn);
    _Float16* t = hp; hp = hn; hn = t;   // hp now holds latest h (f16)
  }

  // Policy head + softmax over actions + output layouts
  policy_gemm_wmma<<<2048, 64, 0, stream>>>(hp, pol16, lg32);
  softmax_out_kernel<<<512, 256, 0, stream>>>(lg32, out_logits, out_probs);
  hk_out_kernel<<<16384, 256, 0, stream>>>(h32, out_hk);
}
